// DirectedEdgeAttentionLayer_57019985822556
// MI455X (gfx1250) — matrix operand order repaired and checked by reading, not compiled, not run on hardware
//
#include <hip/hip_runtime.h>
#include <hip/hip_bf16.h>

// Problem constants (match reference)
#define NN       50000
#define NE       800000
#define NDIM     128
#define EDIM     64
#define HID      128
#define NHEADS   4
#define HDIM     32

typedef __attribute__((ext_vector_type(16))) _Float16 v16h;
typedef __attribute__((ext_vector_type(8)))  _Float16 v8h;
typedef __attribute__((ext_vector_type(8)))  float    v8f;

// ------------------------------------------------------------------
// Utility kernels
// ------------------------------------------------------------------
__global__ void k_zero_f32(float* __restrict__ p, long n) {
  long t = (long)blockIdx.x * blockDim.x + threadIdx.x;
  if (t < n) p[t] = 0.0f;
}

__global__ void k_cvt_f16(const float* __restrict__ in, _Float16* __restrict__ out, long n) {
  long t = (long)blockIdx.x * blockDim.x + threadIdx.x;
  if (t < n) out[t] = (_Float16)in[t];
}

// W is (K x N) row-major fp32 -> Wt is (N x K) row-major f16 (column-major W)
__global__ void k_transpose_cvt(const float* __restrict__ W, _Float16* __restrict__ Wt,
                                int K, int N) {
  long t = (long)blockIdx.x * blockDim.x + threadIdx.x;
  long total = (long)K * N;
  if (t >= total) return;
  int k = (int)(t / N);
  int n = (int)(t % N);
  Wt[(long)n * K + k] = (_Float16)W[t];
}

__global__ void k_concat4(const float* __restrict__ a, const float* __restrict__ b,
                          const float* __restrict__ c, const float* __restrict__ d,
                          float* __restrict__ out, int len) {
  int t = blockIdx.x * blockDim.x + threadIdx.x;
  if (t >= 4 * len) return;
  int which = t / len, i = t % len;
  const float* src = (which == 0) ? a : (which == 1) ? b : (which == 2) ? c : d;
  out[t] = src[i];
}

// ------------------------------------------------------------------
// Fragment helpers (CDNA5 WMMA 16x16x32 f16 layouts, wave32)
//   A 16x32: lane&15 = M row; lanes<16 hold K {0..15}, lanes>=16 {16..31} (contiguous, same as B)
//   B 32x16 (column-major source): lane = (k/16)*16 + n; 16 contiguous K per lane
//   C/D: lane&15 = N col; VGPR r -> M = r + 8*(lane>=16)
// ------------------------------------------------------------------
__device__ __forceinline__ v16h make_afrag(v8h lo, v8h hi) {
  v16h a;
#pragma unroll
  for (int i = 0; i < 8; ++i) { a[i] = lo[i]; a[8 + i] = hi[i]; }
  return a;
}

// ------------------------------------------------------------------
// Multi-N-tile WMMA GEMM: C(MxNcols) = A(MxK, lda) * B + bias.
// Bt is (Ncols x K, ldb) row-major f16. Each wave computes a 16 x (16*NT)
// strip: the A fragment is loaded ONCE per k-step and reused across all
// NT tiles (cuts A HBM traffic by NT; B panels are L2-resident).
// ACT: 0=none, 1=relu, 2=sigmoid. Tile predicate is wave-uniform ->
// EXEC stays all-ones around WMMA.
// ------------------------------------------------------------------
template <int NT, int ACT, bool OUT_F16>
__global__ __launch_bounds__(256) void k_wmma_gemm(
    const _Float16* __restrict__ A, int lda,
    const _Float16* __restrict__ Bt, int ldb,
    const float* __restrict__ bias,
    float* __restrict__ Cf, _Float16* __restrict__ Ch, int ldc,
    int M, int Ncols, int K)
{
  const int wave = threadIdx.x >> 5;
  const int lane = threadIdx.x & 31;
  const int ngroups = Ncols / (16 * NT);
  const long numStrips = (long)(M >> 4) * ngroups;
  const long strip = (long)blockIdx.x * 8 + wave;
  if (strip >= numStrips) return;                  // uniform per wave

  const int m0 = (int)(strip / ngroups) << 4;
  const int n0 = (int)(strip % ngroups) * (16 * NT);
  const int l15   = lane & 15;
  const int abase = (lane < 16) ? 0 : 16;
  const int bbase = (lane < 16) ? 0 : 16;

  const _Float16* __restrict__ Arow = A + (long)(m0 + l15) * lda;

  v8f zero = {};
  v8f acc[NT];
#pragma unroll
  for (int t = 0; t < NT; ++t) acc[t] = zero;

  for (int kk = 0; kk < K; kk += 32) {
    __builtin_prefetch((const void*)(Arow + kk + 64), 0, 3);   // global_prefetch_b8
    v8h alo = *(const v8h*)(Arow + kk + abase);
    v8h ahi = *(const v8h*)(Arow + kk + 8 + abase);
    v16h a = make_afrag(alo, ahi);
#pragma unroll
    for (int t = 0; t < NT; ++t) {
      const _Float16* Bcol = Bt + (long)(n0 + t * 16 + l15) * ldb;
      v16h b = *(const v16h*)(Bcol + kk + bbase);
      acc[t] = __builtin_amdgcn_wmma_f32_16x16x32_f16(
          false, a, false, b, (short)0, acc[t], false, false);
    }
  }

#pragma unroll
  for (int t = 0; t < NT; ++t) {
    const int ncol = n0 + t * 16 + l15;
    const float bv = bias ? bias[ncol] : 0.0f;
#pragma unroll
    for (int r = 0; r < 8; ++r) {
      int orow = m0 + r + ((lane < 16) ? 0 : 8);
      float v = acc[t][r] + bv;
      if (ACT == 1) v = fmaxf(v, 0.0f);
      else if (ACT == 2) v = 1.0f / (1.0f + __expf(-v));
      if (OUT_F16) Ch[(long)orow * ldc + ncol] = (_Float16)v;
      else         Cf[(long)orow * ldc + ncol] = v;
    }
  }
}

// ------------------------------------------------------------------
// Fused edge MLP: eu = relu([node_out[src]|node_out[dst]|edge_feats] @ Wu1
//                           + bu1) @ Wu2 + bu2
// The concat gather is folded into the A-fragment load (node_out is
// L2-resident, 25.6MB); h1 lives only in a 4KB per-wave LDS tile.
// Segment boundaries (128/256/320) are all multiples of 8 and fragment
// runs are 8-aligned, so a run never straddles segments; the segment
// select is branchless (pointer cndmask), keeping EXEC all-ones.
// ------------------------------------------------------------------
__device__ __forceinline__ void gather_cat8(
    const float* __restrict__ node_out, const float* __restrict__ edge_feats,
    long s, long d, long e, int k, _Float16* out8)
{
  const float* p = (k < 128) ? (node_out + s * 128 + k)
                 : (k < 256) ? (node_out + d * 128 + (k - 128))
                             : (edge_feats + e * 64 + (k - 256));
  float4 x0 = ((const float4*)p)[0];
  float4 x1 = ((const float4*)p)[1];
  out8[0] = (_Float16)x0.x; out8[1] = (_Float16)x0.y;
  out8[2] = (_Float16)x0.z; out8[3] = (_Float16)x0.w;
  out8[4] = (_Float16)x1.x; out8[5] = (_Float16)x1.y;
  out8[6] = (_Float16)x1.z; out8[7] = (_Float16)x1.w;
}

__global__ __launch_bounds__(256) void k_edge_mlp_fused(
    const float* __restrict__ node_out, const float* __restrict__ edge_feats,
    const int* __restrict__ srcI, const int* __restrict__ dstI,
    const _Float16* __restrict__ Wu1_t /*128x320*/, const float* __restrict__ bu1,
    const _Float16* __restrict__ Wu2_t /*64x128*/,  const float* __restrict__ bu2,
    float* __restrict__ eu /*NE x 64*/)
{
  __shared__ _Float16 lds[8][16 * 128];            // 8 waves x 4KB h1 tile
  const int wave = threadIdx.x >> 5;
  const int lane = threadIdx.x & 31;
  const long tile = (long)blockIdx.x * 8 + wave;   // 16 edges per tile
  if (tile >= (long)(NE / 16)) return;             // uniform per wave

  const int  l15   = lane & 15;
  const int  hi8   = (lane < 16) ? 0 : 8;
  const int  abase = (lane < 16) ? 0 : 16;
  const int  bbase = (lane < 16) ? 0 : 16;
  const long e = tile * 16 + l15;
  const long s = srcI[e];
  const long d = dstI[e];

  v8f zero = {};

  // ---- MLP1: h1(16x128) = relu(cat @ Wu1 + bu1), K = 320 ----
  v8f acc1[8];
#pragma unroll
  for (int t = 0; t < 8; ++t) acc1[t] = zero;

#pragma unroll
  for (int kk = 0; kk < 320; kk += 32) {
    _Float16 af[16];
    gather_cat8(node_out, edge_feats, s, d, e, kk + abase,      af);
    gather_cat8(node_out, edge_feats, s, d, e, kk + 8 + abase, af + 8);
    v16h a;
#pragma unroll
    for (int i = 0; i < 16; ++i) a[i] = af[i];
#pragma unroll
    for (int t = 0; t < 8; ++t) {
      const _Float16* Bcol = Wu1_t + (long)(t * 16 + l15) * 320;
      v16h b = *(const v16h*)(Bcol + kk + bbase);
      acc1[t] = __builtin_amdgcn_wmma_f32_16x16x32_f16(
          false, a, false, b, (short)0, acc1[t], false, false);
    }
  }

  // bias + relu -> f16, spill C-layout into row-major LDS tile
#pragma unroll
  for (int t = 0; t < 8; ++t) {
    const int col = t * 16 + l15;
    const float bv = bu1[col];
#pragma unroll
    for (int r = 0; r < 8; ++r) {
      const int row = r + hi8;
      lds[wave][row * 128 + col] = (_Float16)fmaxf(acc1[t][r] + bv, 0.0f);
    }
  }
  // Within-wave DS ordering: no barrier needed (private LDS slice,
  // compiler inserts s_wait_dscnt before dependent ds_loads).

  // ---- MLP2: eu(16x64) = h1 @ Wu2 + bu2, K = 128 ----
  v8f acc2[4];
#pragma unroll
  for (int t = 0; t < 4; ++t) acc2[t] = zero;

#pragma unroll
  for (int kk = 0; kk < 128; kk += 32) {
    v8h alo = *(const v8h*)&lds[wave][l15 * 128 + kk + abase];
    v8h ahi = *(const v8h*)&lds[wave][l15 * 128 + kk + 8 + abase];
    v16h a = make_afrag(alo, ahi);
#pragma unroll
    for (int t = 0; t < 4; ++t) {
      const _Float16* Bcol = Wu2_t + (long)(t * 16 + l15) * 128;
      v16h b = *(const v16h*)(Bcol + kk + bbase);
      acc2[t] = __builtin_amdgcn_wmma_f32_16x16x32_f16(
          false, a, false, b, (short)0, acc2[t], false, false);
    }
  }

#pragma unroll
  for (int t = 0; t < 4; ++t) {
    const int col = t * 16 + l15;
    const float bv = bu2[col];
#pragma unroll
    for (int r = 0; r < 8; ++r) {
      const long orow = tile * 16 + r + hi8;
      eu[orow * 64 + col] = acc2[t][r] + bv;
    }
  }
}

// ------------------------------------------------------------------
// Attention: scores + segment max (bit-pattern atomicMax; init 0.0f
// matches reference's max-with-zero semantics)
// ------------------------------------------------------------------
__global__ void k_attn_scores(const float* __restrict__ QKVO, const float* __restrict__ Ep,
                              const int* __restrict__ srcI, const int* __restrict__ dstI,
                              float* __restrict__ attn, int* __restrict__ node_max_bits,
                              long total /*E*4*/) {
  long t = (long)blockIdx.x * blockDim.x + threadIdx.x;
  if (t >= total) return;
  long e = t >> 2; int h = (int)(t & 3);
  int s = srcI[e], d = dstI[e];
  const float4* q  = (const float4*)(QKVO + (long)d * 512 + h * HDIM);
  const float4* kv = (const float4*)(QKVO + (long)s * 512 + 128 + h * HDIM);
  const float4* ep = (const float4*)(Ep + (long)e * HID + h * HDIM);
  float acc = 0.0f;
#pragma unroll
  for (int i = 0; i < 8; ++i) {
    float4 qq = q[i], kk = kv[i], ee = ep[i];
    acc += qq.x * (kk.x + ee.x) + qq.y * (kk.y + ee.y)
         + qq.z * (kk.z + ee.z) + qq.w * (kk.w + ee.w);
  }
  acc *= 0.17677669529663687f;                    // 1/sqrt(32)
  attn[t] = acc;
  // positives order correctly as ints; negatives never beat 0-bits init
  atomicMax(node_max_bits + (long)d * 4 + h, __float_as_int(acc));
}

__global__ void k_attn_exp(const float* __restrict__ attn, const int* __restrict__ node_max_bits,
                           const int* __restrict__ dstI, float* __restrict__ wexp,
                           float* __restrict__ node_sum, long total /*E*4*/) {
  long t = (long)blockIdx.x * blockDim.x + threadIdx.x;
  if (t >= total) return;
  long e = t >> 2; int h = (int)(t & 3);
  int d = dstI[e];
  float m = __int_as_float(node_max_bits[(long)d * 4 + h]);
  float w = __expf(attn[t] - m);
  wexp[t] = w;
  atomicAdd(node_sum + (long)d * 4 + h, w);
}

__global__ void k_attn_agg(const float* __restrict__ wexp, const float* __restrict__ node_sum,
                           const float* __restrict__ QKVO,
                           const int* __restrict__ srcI, const int* __restrict__ dstI,
                           float* __restrict__ agg, long total /*E*128*/) {
  long t = (long)blockIdx.x * blockDim.x + threadIdx.x;
  if (t >= total) return;
  long e = t >> 7; int c = (int)(t & 127); int h = c >> 5;
  int s = srcI[e], d = dstI[e];
  float w = wexp[e * 4 + h] / (node_sum[(long)d * 4 + h] + 1e-10f);
  float v = QKVO[(long)s * 512 + 256 + c];        // V block
  atomicAdd(agg + (long)d * 128 + c, w * v);      // agg is L2-resident (25.6MB)
}

// cat16 = f16([agg | o_node])  (NN x 256)
__global__ void k_build_cat16(const float* __restrict__ agg, const float* __restrict__ QKVO,
                              _Float16* __restrict__ cat, long total /*NN*256*/) {
  long t = (long)blockIdx.x * blockDim.x + threadIdx.x;
  if (t >= total) return;
  long n = t >> 8; int c = (int)(t & 255);
  float v = (c < 128) ? agg[n * 128 + c] : QKVO[n * 512 + 384 + (c - 128)];
  cat[t] = (_Float16)v;
}

// ------------------------------------------------------------------
// Node finalize: gated residual + layernorm(128)
// ------------------------------------------------------------------
__global__ __launch_bounds__(128) void k_node_finalize(
    const float* __restrict__ node_feats, const float* __restrict__ g,
    const float* __restrict__ aggWo, const float* __restrict__ lng,
    const float* __restrict__ lnb, float* __restrict__ node_out) {
  int n = blockIdx.x, t = threadIdx.x;
  float x0 = node_feats[(long)n * 128 + t];
  float gv = g[(long)n * 128 + t];
  float aw = aggWo[(long)n * 128 + t];
  float x = x0 + (gv * aw + (1.0f - gv) * x0);    // residual + update
  __shared__ float sm[128];
  sm[t] = x; __syncthreads();
  for (int s = 64; s > 0; s >>= 1) { if (t < s) sm[t] += sm[t + s]; __syncthreads(); }
  float mean = sm[0] * (1.0f / 128.0f); __syncthreads();
  float dx = x - mean;
  sm[t] = dx * dx; __syncthreads();
  for (int s = 64; s > 0; s >>= 1) { if (t < s) sm[t] += sm[t + s]; __syncthreads(); }
  float var = sm[0] * (1.0f / 128.0f);
  node_out[(long)n * 128 + t] = dx * rsqrtf(var + 1e-5f) * lng[t] + lnb[t];
}

// Edge finalize: residual + layernorm(64)
__global__ __launch_bounds__(64) void k_edge_finalize(
    const float* __restrict__ edge_feats, const float* __restrict__ eu,
    const float* __restrict__ lng, const float* __restrict__ lnb,
    float* __restrict__ edge_out) {
  long e = blockIdx.x; int t = threadIdx.x;
  float x = edge_feats[e * 64 + t] + eu[e * 64 + t];
  __shared__ float sm[64];
  sm[t] = x; __syncthreads();
  for (int s = 32; s > 0; s >>= 1) { if (t < s) sm[t] += sm[t + s]; __syncthreads(); }
  float mean = sm[0] * (1.0f / 64.0f); __syncthreads();
  float dx = x - mean;
  sm[t] = dx * dx; __syncthreads();
  for (int s = 32; s > 0; s >>= 1) { if (t < s) sm[t] += sm[t + s]; __syncthreads(); }
  float var = sm[0] * (1.0f / 64.0f);
  edge_out[e * 64 + t] = dx * rsqrtf(var + 1e-5f) * lng[t] + lnb[t];
}

// ------------------------------------------------------------------
// Host-side orchestration
// ------------------------------------------------------------------
static inline unsigned blocks_for(long n, int bs) { return (unsigned)((n + bs - 1) / bs); }

extern "C" void kernel_launch(void* const* d_in, const int* in_sizes, int n_in,
                              void* d_out, int out_size, void* d_ws, size_t ws_size,
                              hipStream_t stream) {
  const float* node_feats = (const float*)d_in[0];
  const float* edge_feats = (const float*)d_in[1];
  const int*   edge_index = (const int*)d_in[2];
  const float* Wq = (const float*)d_in[3];  const float* bq = (const float*)d_in[4];
  const float* Wk = (const float*)d_in[5];  const float* bk = (const float*)d_in[6];
  const float* Wv = (const float*)d_in[7];  const float* bv = (const float*)d_in[8];
  const float* We = (const float*)d_in[9];  const float* be = (const float*)d_in[10];
  const float* Wg = (const float*)d_in[11]; const float* bg = (const float*)d_in[12];
  const float* Wo = (const float*)d_in[13]; const float* bo = (const float*)d_in[14];
  const float* Wu1 = (const float*)d_in[15]; const float* bu1 = (const float*)d_in[16];
  const float* Wu2 = (const float*)d_in[17]; const float* bu2 = (const float*)d_in[18];
  const float* ln1g = (const float*)d_in[19]; const float* ln1b = (const float*)d_in[20];
  const float* ln2g = (const float*)d_in[21]; const float* ln2b = (const float*)d_in[22];

  const int* srcI = edge_index;
  const int* dstI = edge_index + NE;

  float* node_out = (float*)d_out;                       // NN x 128
  float* edge_out = (float*)d_out + (long)NN * NDIM;     // NE x 64

  // ---- workspace bump allocator (256-byte aligned) ----
  char* ws = (char*)d_ws; size_t off = 0;
  auto alloc = [&](size_t bytes) -> void* {
    off = (off + 255) & ~(size_t)255;
    void* p = ws + off; off += bytes; return p;
  };
  _Float16* nf16     = (_Float16*)alloc((size_t)NN * NDIM * 2);
  _Float16* ef16     = (_Float16*)alloc((size_t)NE * EDIM * 2);
  _Float16* Wqkvo_t  = (_Float16*)alloc((size_t)512 * 128 * 2);   // [Wq^T|Wk^T|Wv^T|Wo^T]
  float*    bqkvo    = (float*)   alloc((size_t)512 * 4);
  _Float16* We_t     = (_Float16*)alloc((size_t)128 * 64 * 2);
  _Float16* Wg_t     = (_Float16*)alloc((size_t)128 * 256 * 2);
  _Float16* Wu1_t    = (_Float16*)alloc((size_t)128 * 320 * 2);
  _Float16* Wu2_t    = (_Float16*)alloc((size_t)64 * 128 * 2);
  float*    QKVO     = (float*)   alloc((size_t)NN * 512 * 4);    // 102.4 MB
  float*    Ep       = (float*)   alloc((size_t)NE * HID * 4);    // 409.6 MB
  float*    attn     = (float*)   alloc((size_t)NE * 4 * 4);
  float*    wexp     = (float*)   alloc((size_t)NE * 4 * 4);
  int*      nmax     = (int*)     alloc((size_t)NN * 4 * 4);
  float*    nsum     = (float*)   alloc((size_t)NN * 4 * 4);
  float*    agg      = (float*)   alloc((size_t)NN * 128 * 4);
  _Float16* cat16    = (_Float16*)alloc((size_t)NN * 256 * 2);
  float*    gbuf     = (float*)   alloc((size_t)NN * 128 * 4);
  float*    aggWo    = (float*)   alloc((size_t)NN * 128 * 4);
  float*    eu       = (float*)   alloc((size_t)NE * 64 * 4);

  // ---- stage 0: conversions / weight packing ----
  k_cvt_f16<<<blocks_for((long)NN * NDIM, 256), 256, 0, stream>>>(node_feats, nf16, (long)NN * NDIM);
  k_cvt_f16<<<blocks_for((long)NE * EDIM, 256), 256, 0, stream>>>(edge_feats, ef16, (long)NE * EDIM);
  k_transpose_cvt<<<blocks_for(128 * 128, 256), 256, 0, stream>>>(Wq, Wqkvo_t + 0 * 128 * 128, 128, 128);
  k_transpose_cvt<<<blocks_for(128 * 128, 256), 256, 0, stream>>>(Wk, Wqkvo_t + 1 * 128 * 128, 128, 128);
  k_transpose_cvt<<<blocks_for(128 * 128, 256), 256, 0, stream>>>(Wv, Wqkvo_t + 2 * 128 * 128, 128, 128);
  k_transpose_cvt<<<blocks_for(128 * 128, 256), 256, 0, stream>>>(Wo, Wqkvo_t + 3 * 128 * 128, 128, 128);
  k_concat4<<<blocks_for(512, 256), 256, 0, stream>>>(bq, bk, bv, bo, bqkvo, 128);
  k_transpose_cvt<<<blocks_for(64 * 128, 256), 256, 0, stream>>>(We, We_t, 64, 128);
  k_transpose_cvt<<<blocks_for(256 * 128, 256), 256, 0, stream>>>(Wg, Wg_t, 256, 128);
  k_transpose_cvt<<<blocks_for(320 * 128, 256), 256, 0, stream>>>(Wu1, Wu1_t, 320, 128);
  k_transpose_cvt<<<blocks_for(128 * 64, 256), 256, 0, stream>>>(Wu2, Wu2_t, 128, 64);

  // ---- stage 1: fused QKV+o_node GEMM (NN x 128) @ (128 x 512), 8 N-tiles/wave ----
  {
    long strips = ((long)NN / 16) * (512 / 128);
    k_wmma_gemm<8, 0, false><<<blocks_for(strips, 8), 256, 0, stream>>>(
        nf16, NDIM, Wqkvo_t, 128, bqkvo, QKVO, nullptr, 512, NN, 512, 128);
  }
  // ---- stage 2: edge projection Ep = edge_feats @ We + be  (NE x 128) ----
  {
    long strips = ((long)NE / 16) * (HID / 128);
    k_wmma_gemm<8, 0, false><<<blocks_for(strips, 8), 256, 0, stream>>>(
        ef16, EDIM, We_t, EDIM, be, Ep, nullptr, HID, NE, HID, EDIM);
  }

  // ---- stage 3: attention (scores -> segment softmax -> scatter V) ----
  k_zero_f32<<<blocks_for((long)NN * 4, 256), 256, 0, stream>>>((float*)nmax, (long)NN * 4);  // bits(0.0f)==0
  k_zero_f32<<<blocks_for((long)NN * 4, 256), 256, 0, stream>>>(nsum, (long)NN * 4);
  k_zero_f32<<<blocks_for((long)NN * 128, 256), 256, 0, stream>>>(agg, (long)NN * 128);
  k_attn_scores<<<blocks_for((long)NE * 4, 256), 256, 0, stream>>>(QKVO, Ep, srcI, dstI, attn, nmax, (long)NE * 4);
  k_attn_exp<<<blocks_for((long)NE * 4, 256), 256, 0, stream>>>(attn, nmax, dstI, wexp, nsum, (long)NE * 4);
  k_attn_agg<<<blocks_for((long)NE * 128, 256), 256, 0, stream>>>(wexp, nsum, QKVO, srcI, dstI, agg, (long)NE * 128);

  // ---- stage 4: gate + node update ----
  k_build_cat16<<<blocks_for((long)NN * 256, 256), 256, 0, stream>>>(agg, QKVO, cat16, (long)NN * 256);
  {
    long strips = ((long)NN / 16);
    // g = sigmoid([agg|o_node] @ Wg + bg)
    k_wmma_gemm<8, 2, false><<<blocks_for(strips, 8), 256, 0, stream>>>(
        cat16, 256, Wg_t, 256, bg, gbuf, nullptr, HID, NN, HID, 256);
    // aggWo = agg @ Wo + bo   (A = first 128 cols of cat16, lda=256)
    k_wmma_gemm<8, 0, false><<<blocks_for(strips, 8), 256, 0, stream>>>(
        cat16, 256, Wqkvo_t + 3 * 128 * 128, 128, bo, aggWo, nullptr, HID, NN, HID, 128);
  }
  k_node_finalize<<<NN, 128, 0, stream>>>(node_feats, gbuf, aggWo, ln1g, ln1b, node_out);

  // ---- stage 5: fused edge MLP (gather + MLP1 + LDS h1 tile + MLP2) ----
  k_edge_mlp_fused<<<blocks_for((long)(NE / 16), 8), 256, 0, stream>>>(
      node_out, edge_feats, srcI, dstI, Wu1_t, bu1, Wu2_t, bu2, eu);
  k_edge_finalize<<<NE, 64, 0, stream>>>(edge_feats, eu, ln2g, ln2b, edge_out);
}
